// ResidualWaveFrobinator_64115271794987
// MI455X (gfx1250) — matrix operand order repaired and checked by reading, not compiled, not run on hardware
//
#include <hip/hip_runtime.h>
#include <stdint.h>

// ---------------------------------------------------------------------------
// ResidualWaveFrobinator on gfx1250 (MI455X, wave32, WMMA)
//
// Dominant math rewritten as one GEMM  Z[8192 x 2176] * M[2176 x 128]
// executed with v_wmma_f32_16x16x32_bf16 (f32 accumulate), Z built on the fly.
// ---------------------------------------------------------------------------

typedef __bf16 bf16_t;
typedef bf16_t v16bf __attribute__((ext_vector_type(16)));
typedef float  v8f   __attribute__((ext_vector_type(8)));
typedef unsigned int v4u __attribute__((ext_vector_type(4)));
typedef unsigned int v8u __attribute__((ext_vector_type(8)));

#define FIN   128
#define FH    16
#define HW    4096   // 64*64
#define NPIX  8192   // B*H*W
#define KTOT  2176   // 128*16 + 128
#define KSTEPS 68    // KTOT / 32

static __device__ __forceinline__ unsigned short f2bf(float f) {
  unsigned int u = __float_as_uint(f);
  unsigned int r = u + 0x7FFFu + ((u >> 16) & 1u);   // RNE
  return (unsigned short)(r >> 16);
}

static __device__ __forceinline__ v16bf load_frag16(const unsigned short* p) {
  // two runs of 8 bf16 (p[0..7], p[16..23]) from LDS, 16B-aligned: two b128 loads
  v4u a = *(const v4u*)(p);
  v4u b = *(const v4u*)(p + 16);
  v8u w = __builtin_shufflevector(a, b, 0, 1, 2, 3, 4, 5, 6, 7);
  return __builtin_bit_cast(v16bf, w);
}

// ---------------------------------------------------------------------------
// Kernel 1: per-pixel prep. 64 pixels/block, 256 threads.
//   a = wa.x+ba, b = wb.x+bb ; fa=[sin a;cos a], fb=[sin b;cos b]
//   c[j] = bc[j] + (1/16) * fa^T WC_j fb        (WC_j = wc[j] as 32x32)
//   writes c to ws_c, and per-block (sum,sumsq) partials for instance-norm.
// ---------------------------------------------------------------------------
__global__ __launch_bounds__(256) void prep_kernel(
    const float* __restrict__ x, const float* __restrict__ wa, const float* __restrict__ ba,
    const float* __restrict__ wb, const float* __restrict__ bb,
    const float* __restrict__ wc, const float* __restrict__ bc,
    float* __restrict__ ws_c, float* __restrict__ ws_part)
{
  __shared__ float xt[64 * 129];    // [pixel][f], padded row (bank-safe scatter fill)
  __shared__ float fafb[64 * 64];   // [pixel][fa(32) | fb(32)]
  __shared__ float carr[64 * 16];   // [pixel][j]

  const int tid = threadIdx.x;
  const int pixbase = blockIdx.x * 64;          // global pixel base (one image: 64 | 4096)
  const int b = pixbase >> 12;
  const int ploc = pixbase & (HW - 1);
  const float* xb = x + (size_t)b * (FIN * HW) + ploc;   // xb[f*HW + p]

  for (int idx = tid; idx < 64 * 128; idx += 256) {
    int f = idx >> 6, p = idx & 63;
    xt[p * 129 + f] = xb[f * HW + p];           // coalesced global read
  }
  __syncthreads();

  // Phase A: 64 px * 32 dots; one wave = one pixel -> LDS broadcast reads
  for (int task = tid; task < 64 * 32; task += 256) {
    int p = task >> 5, i = task & 31;
    const float* wrow = (i < 16) ? (wa + i * FIN) : (wb + (i - 16) * FIN);
    float acc = (i < 16) ? ba[i] : bb[i - 16];
    const float* xr = xt + p * 129;
    for (int f = 0; f < FIN; ++f) acc = fmaf(wrow[f], xr[f], acc);
    float s = __sinf(acc), c = __cosf(acc);
    int off = (i < 16) ? 0 : 32;
    int ii = i & 15;
    fafb[p * 64 + off + ii] = s;
    fafb[p * 64 + off + 16 + ii] = c;
  }
  __syncthreads();

  // Phase B: c[p][j] = bc[j] + (1/16) * sum_{pp,q} wc[j,32*pp+q] fa[pp] fb[q]
  for (int task = tid; task < 64 * 16; task += 256) {
    int p = task >> 4, j = task & 15;
    const float* fr = fafb + p * 64;            // fa = fr[0..31], fb = fr[32..63]
    const float* wr = wc + j * 1024;
    float acc = 0.f;
    for (int pp = 0; pp < 32; ++pp) {
      const float* wq = wr + pp * 32;
      float inner = 0.f;
      for (int q = 0; q < 32; ++q) inner = fmaf(wq[q], fr[32 + q], inner);
      acc = fmaf(fr[pp], inner, acc);
    }
    float cc = bc[j] + acc * (1.f / 16.f);
    ws_c[(size_t)(pixbase + p) * FH + j] = cc;
    carr[p * 16 + j] = cc;
  }
  __syncthreads();

  // Deterministic per-block partial stats (fixed summation order)
  if (tid < 16) {
    float s = 0.f, s2 = 0.f;
    for (int p = 0; p < 64; ++p) { float v = carr[p * 16 + tid]; s += v; s2 = fmaf(v, v, s2); }
    ws_part[blockIdx.x * 32 + tid * 2 + 0] = s;
    ws_part[blockIdx.x * 32 + tid * 2 + 1] = s2;
  }
}

// ---------------------------------------------------------------------------
// Kernel 2: finalize instance-norm mu / rstd per (b, j). 32 threads.
// ---------------------------------------------------------------------------
__global__ void norm_kernel(const float* __restrict__ ws_part, float* __restrict__ ws_norm)
{
  int t = threadIdx.x;
  if (t >= 32) return;
  int b = t >> 4, j = t & 15;
  float s = 0.f, s2 = 0.f;
  for (int blk = 0; blk < 64; ++blk) {          // 64 blocks per image, fixed order
    const float* pp = ws_part + ((b * 64 + blk) * 32 + j * 2);
    s += pp[0]; s2 += pp[1];
  }
  float mu  = s * (1.f / 4096.f);
  float var = s2 * (1.f / 4096.f) - mu * mu;
  float rstd = rsqrtf(var + 1e-5f);
  ws_norm[(b * 16 + j) * 2 + 0] = mu;
  ws_norm[(b * 16 + j) * 2 + 1] = rstd;
}

// ---------------------------------------------------------------------------
// Kernel 3: build B matrix (bf16) in WMMA-friendly layout:
//   wsB[(t*128 + n)*32 + kj]  for K-step t, output column n, K-in-step kj.
//   kk = 32t+kj :  kk<2048 -> wl[((kk>>4)*128 + n)*16 + (kk&15)] / sqrt(128)
//                  else    -> bl[(kk-2048)*128 + n] / sqrt(128)       (bias rows)
// ---------------------------------------------------------------------------
__global__ __launch_bounds__(256) void bmat_kernel(
    const float* __restrict__ wl, const float* __restrict__ bl, unsigned short* __restrict__ wsB)
{
  int idx = blockIdx.x * 256 + threadIdx.x;
  if (idx >= KSTEPS * 128 * 32) return;
  int t = idx >> 12;
  int rem = idx & 4095;
  int n = rem >> 5, kj = rem & 31;
  int kk = t * 32 + kj;
  float v;
  if (kk < 2048) v = wl[((size_t)(kk >> 4) * 128 + n) * 16 + (kk & 15)];
  else           v = bl[(size_t)(kk - 2048) * 128 + n];
  wsB[idx] = f2bf(v * 0.08838834764831845f);    // 1/sqrt(128)
}

// ---------------------------------------------------------------------------
// Kernel 4: fused WMMA GEMM + residual.
//   64 blocks * 8 waves; block = 128 pixels x 128 outputs; wave = 16 px x 128 out.
//   68 K-steps of v_wmma_f32_16x16x32_bf16 (8 N-tiles each), A built on the fly.
// ---------------------------------------------------------------------------
#define RSX 136   // Xs row stride (bf16): 128 + 8 pad
#define RSB 40    // Bs row stride (bf16): 32 + 8 pad (keeps 16B alignment)
#define RSY 129   // Ys row stride (f32)

__global__ __launch_bounds__(256) void gemm_kernel(
    const float* __restrict__ x, const unsigned short* __restrict__ wsB,
    const float* __restrict__ ws_c, const float* __restrict__ ws_norm,
    float* __restrict__ out)
{
  __shared__ unsigned short Xs[128 * RSX];      // x tile, bf16 [pixel][f]
  __shared__ unsigned short Bs[2][128 * RSB];   // double-buffered B tile [n][K]
  __shared__ float Ys[128 * RSY];               // y tile for coalesced transpose-out

  const int tid  = threadIdx.x;
  const int lane = tid & 31;
  const int wv   = tid >> 5;                    // wave id 0..7 -> pixels wv*16..wv*16+15
  const int mloc = lane & 15;                   // M (pixel row) within tile
  const int koff = (lane >> 4) << 3;            // A/fc K half select: 0 or 8

  const int pixbase = blockIdx.x * 128;
  const int b    = pixbase >> 12;
  const int ploc = pixbase & (HW - 1);
  const float* xb = x + (size_t)b * (FIN * HW) + ploc;

  // stage x tile as bf16 [pixel][channel]
  for (int idx = tid; idx < 128 * 128; idx += 256) {
    int f = idx >> 7, p = idx & 127;
    Xs[p * RSX + f] = f2bf(xb[f * HW + p]);     // coalesced read, bank-safe scatter
  }
  // stage first B tile (8KB): 128 rows x 2 halves, one 32B chunk per thread
  {
    int n = tid >> 1, half = tid & 1;
    const v4u* src = (const v4u*)(wsB + ((size_t)0 * 128 + n) * 32 + half * 16);
    v4u* dst = (v4u*)(&Bs[0][n * RSB + half * 16]);
    dst[0] = src[0]; dst[1] = src[1];
  }
  // preload this lane's 8 normalized fc values (constant over the K loop)
  float fcl[8];
  {
    const float* cp = ws_c + (size_t)(pixbase + wv * 16 + mloc) * FH + koff;
    const float* np = ws_norm + (b * 16 + koff) * 2;
#pragma unroll
    for (int k = 0; k < 8; ++k) fcl[k] = (cp[k] - np[k * 2]) * np[k * 2 + 1];
  }
  v8f acc[8] = {};
  __syncthreads();

  const unsigned short* xrow = Xs + (wv * 16 + mloc) * RSX;

  for (int t = 0; t < KSTEPS; ++t) {
    const int buf = t & 1;
    if (t + 1 < KSTEPS) {                       // stage next B tile into other buffer
      int n = tid >> 1, half = tid & 1;
      const v4u* src = (const v4u*)(wsB + ((size_t)(t + 1) * 128 + n) * 32 + half * 16);
      v4u s0 = src[0], s1 = src[1];
      v4u* dst = (v4u*)(&Bs[buf ^ 1][n * RSB + half * 16]);
      dst[0] = s0; dst[1] = s1;
    }
    if (t + 2 < KSTEPS)                         // -> global_prefetch_b8
      __builtin_prefetch(wsB + (size_t)(t + 2) * 4096 + (tid << 4), 0, 1);

    // ---- build A fragment (16x32 bf16, ISA 7.12.2 layout) ----
    v16bf av;
    if (t < 64) {
      // Z[n, f*16+k] = x[n,f] * fc[n,k]; this step covers f0=2t, f0+1, k=0..15
      const int f0 = t << 1;
      unsigned int xp = *(const unsigned int*)(xrow + f0);   // bf16 pair (f0, f0+1)
      float x0 = __uint_as_float(xp << 16);
      float x1 = __uint_as_float(xp & 0xFFFF0000u);
#pragma unroll
      for (int i = 0; i < 8; ++i) av[i]     = (bf16_t)(x0 * fcl[i]);
#pragma unroll
      for (int i = 0; i < 8; ++i) av[8 + i] = (bf16_t)(x1 * fcl[i]);
    } else {
      // bias rows: Z[n, 2048+f] = x[n,f]  -> direct b128 fragment loads
      const int base = (t - 64) * 32 + koff;
      av = __builtin_bit_cast(v16bf, __builtin_shufflevector(
             *(const v4u*)(xrow + base), *(const v4u*)(xrow + base + 16),
             0, 1, 2, 3, 4, 5, 6, 7));
    }

    // ---- 8 N-tiles of WMMA ----
    const unsigned short* bbase = Bs[buf];
    const int koffB = (lane >> 4) << 3;         // B frag: K 0..7,16..23 / 8..15,24..31
#pragma unroll
    for (int nt = 0; nt < 8; ++nt) {
      v16bf bv = load_frag16(bbase + (nt * 16 + mloc) * RSB + koffB);
      acc[nt] = __builtin_amdgcn_wmma_f32_16x16x32_bf16(
          false, av, false, bv, (short)0, acc[nt], false, false);
    }
    __syncthreads();
  }

  // write y tiles to LDS (C/D layout: M = r + 8*(lane>=16), N = lane&15)
#pragma unroll
  for (int nt = 0; nt < 8; ++nt) {
#pragma unroll
    for (int r = 0; r < 8; ++r) {
      int prow = wv * 16 + r + ((lane >> 4) << 3);
      Ys[prow * RSY + nt * 16 + mloc] = acc[nt][r];
    }
  }
  __syncthreads();

  // residual + coalesced NCHW store: out[b,o,p] = x[b,o,p] - y[p,o]
  for (int idx = tid; idx < 128 * 128; idx += 256) {
    int o = idx >> 7, p = idx & 127;
    size_t g = (size_t)b * (FIN * HW) + (size_t)o * HW + ploc + p;
    out[g] = x[g] - Ys[p * RSY + o];
  }
}

// ---------------------------------------------------------------------------
extern "C" void kernel_launch(void* const* d_in, const int* in_sizes, int n_in,
                              void* d_out, int out_size, void* d_ws, size_t ws_size,
                              hipStream_t stream) {
  const float* x  = (const float*)d_in[0];
  const float* wa = (const float*)d_in[1];
  const float* ba = (const float*)d_in[2];
  const float* wb = (const float*)d_in[3];
  const float* bb = (const float*)d_in[4];
  const float* wc = (const float*)d_in[5];
  const float* bc = (const float*)d_in[6];
  const float* wl = (const float*)d_in[7];
  const float* bl = (const float*)d_in[8];
  float* out = (float*)d_out;

  // workspace layout (all 16B aligned; total ~1.05 MB)
  float* ws_part = (float*)d_ws;                          // 128*32 f32
  float* ws_norm = ws_part + 128 * 32;                    // 64 f32
  float* ws_c    = ws_norm + 64;                          // 8192*16 f32
  unsigned short* wsB = (unsigned short*)(ws_c + NPIX * FH); // 68*128*32 bf16

  prep_kernel<<<NPIX / 64, 256, 0, stream>>>(x, wa, ba, wb, bb, wc, bc, ws_c, ws_part);
  norm_kernel<<<1, 32, 0, stream>>>(ws_part, ws_norm);
  bmat_kernel<<<(KSTEPS * 128 * 32) / 256, 256, 0, stream>>>(wl, bl, wsB);
  gemm_kernel<<<NPIX / 128, 256, 0, stream>>>(x, wsB, ws_c, ws_norm, out);
}
